// Glo_tkPoolingAttention_49520972923077
// MI455X (gfx1250) — matrix operand order repaired and checked by reading, not compiled, not run on hardware
//
#include <hip/hip_runtime.h>
#include <math.h>

// ---------------- problem constants ----------------
#define BB      8
#define CC      256
#define HH      32
#define WW      32
#define SS      1024          // H*W
#define HEADS   8
#define DH      32
#define QTOT    1025          // 1 global token + 1024 spatial queries
#define QPAD    1040          // padded to multiple of 16
#define NPOOL   1426          // 1024 + 256 + 121 + 25
#define NKCHUNK 45            // ceil(1426/32)
#define NPAD    (NKCHUNK * 32) // 1440: K/V rows padded to whole 32-key chunks
#define SCALEF  0.17677669529663689f  // 32^-0.5
#define EPSF    1e-5f

typedef __attribute__((ext_vector_type(16))) _Float16 v16h;
typedef __attribute__((ext_vector_type(8)))  float    v8f;

__device__ __forceinline__ v8f wmma_f16(v16h a, v16h b, v8f c) {
  // D = A(16x32 f16) * B(32x16 f16) + C(16x16 f32)
  return __builtin_amdgcn_wmma_f32_16x16x32_f16(false, a, false, b, (short)0, c,
                                                false, false);
}

// Fragment layouts per CDNA5 ISA 7.12.2 (wave32):
//  A (16x32 f16, A[m][k]): lane: m=lane&15, h=lane>>4;
//     a[j] = A[m][h*8+j] (j<8),  a[j] = A[m][16+h*8+(j-8)] (j>=8)
//  B (32x16 f16, B[k][n]): lane: n=lane&15, h=lane>>4; same k order as A (this code: k=h*16+j for both A and B)
//  C/D (16x16 f32): c[i] -> M=(lane>>4)*8+i, N=lane&15

// ==================================================================
// 1) global-token init + pad-row zeroing for the Q buffer
// ==================================================================
__global__ void qinit_kernel(const float* __restrict__ gt, _Float16* __restrict__ qf) {
  int tid = blockIdx.x * blockDim.x + threadIdx.x;     // B*HEADS*DH = 2048
  if (tid >= BB * HEADS * DH) return;
  int dh = tid & 31, h = (tid >> 5) & 7, b = tid >> 8;
  _Float16* base = qf + (((size_t)b * HEADS + h) * QPAD) * DH + dh;
  base[0] = (_Float16)gt[dh * HEADS + h];   // glo[head][dh] = gt[dh*HEADS+head]
  for (int r = QTOT; r < QPAD; ++r) base[(size_t)r * DH] = (_Float16)0.f;
}

// zero the padded K/V rows [NPOOL, NPAD) so attention fragments are branch-free
__global__ void kvpad_kernel(_Float16* __restrict__ kf, _Float16* __restrict__ vf) {
  int tid = blockIdx.x * blockDim.x + threadIdx.x;
  const int PR = NPAD - NPOOL;                       // 14
  if (tid >= BB * HEADS * PR * DH) return;
  int dh = tid & 31;
  int r  = (tid >> 5) % PR;
  int bh = tid / (PR * DH);
  size_t idx = ((size_t)bh * NPAD + NPOOL + r) * DH + dh;
  kf[idx] = (_Float16)0.f;
  vf[idx] = (_Float16)0.f;
}

// ==================================================================
// 2) fused QKV projection:  y[b,o,s] = sum_c W[o,c] * x[b,c,s]
//    one wave computes a 64(M) x 16(N) strip: 4 A-fragments share one
//    B-fragment per k-step -> 4x B reuse, 4 wmma per k-step
// ==================================================================
__global__ void qkv_gemm_kernel(const float* __restrict__ x,
                                const float* __restrict__ q_w,
                                const float* __restrict__ kv_w,
                                _Float16* __restrict__ qf,   // [B][H][QPAD][DH]
                                float* __restrict__ k_map,   // [B][C][S]
                                float* __restrict__ v_map) { // [B][C][S]
  const int lane = threadIdx.x & 31;
  const int half = lane >> 4, lm = lane & 15;
  const int n0 = blockIdx.x * 16;   // spatial
  const int m0 = blockIdx.y * 64;   // output channel strip (0..767 in steps of 64)
  const int b  = blockIdx.z;
  const float* xb = x + (size_t)b * CC * SS;

  const float* wrow[4];
#pragma unroll
  for (int t = 0; t < 4; ++t) {
    int m = m0 + t * 16 + lm;
    wrow[t] = (m < CC) ? (q_w + (size_t)m * CC) : (kv_w + (size_t)(m - CC) * CC);
  }
  v8f acc[4] = {{}, {}, {}, {}};
  for (int kk = 0; kk < CC; kk += 32) {
    __builtin_prefetch(xb + (size_t)(kk + 32) * SS + n0, 0, 1);
    v16h bf;                               // B[k][n] = x[b][kk+k][n0+n]
#pragma unroll
    for (int j = 0; j < 16; ++j)
      bf[j] = (_Float16)xb[(size_t)(kk + half * 16 + j) * SS + (n0 + lm)];
#pragma unroll
    for (int t = 0; t < 4; ++t) {
      v16h a;
#pragma unroll
      for (int i = 0; i < 8; ++i) {
        int kb = half * 16 + i * 2;
        a[2 * i]     = (_Float16)wrow[t][kk + kb];
        a[2 * i + 1] = (_Float16)wrow[t][kk + kb + 1];
      }
      acc[t] = wmma_f16(a, bf, acc[t]);
    }
  }
#pragma unroll
  for (int t = 0; t < 4; ++t) {
#pragma unroll
    for (int i = 0; i < 8; ++i) {
      int mo = m0 + t * 16 + half * 8 + i;
      int s  = n0 + lm;
      float v = acc[t][i];
      if (mo < CC) {                      // q: channel = dh*HEADS + head
        int head = mo & (HEADS - 1), dh = mo >> 3;
        qf[(((size_t)b * HEADS + head) * QPAD + (s + 1)) * DH + dh] = (_Float16)v;
      } else if (mo < 2 * CC) {
        k_map[((size_t)b * CC + (mo - CC)) * SS + s] = v;
      } else {
        v_map[((size_t)b * CC + (mo - 2 * CC)) * SS + s] = v;
      }
    }
  }
}

// ==================================================================
// 3) multi-ratio average pooling (exact _pool_mat semantics)
// ==================================================================
__device__ __forceinline__ void pool_region(int n, int& ri, int& p, int& q, int& od) {
  const int dims[4] = {32, 16, 11, 5};
  const int offs[4] = {0, 1024, 1280, 1401};
  ri = (n < 1024) ? 0 : (n < 1280) ? 1 : (n < 1401) ? 2 : 3;
  od = dims[ri];
  int local = n - offs[ri];
  p = local / od; q = local % od;
}

__global__ void pool_kernel(const float* __restrict__ src, float* __restrict__ dst) {
  size_t gid = (size_t)blockIdx.x * blockDim.x + threadIdx.x;
  if (gid >= (size_t)BB * CC * NPOOL) return;
  int n = (int)(gid % NPOOL);
  size_t bc = gid / NPOOL;
  int ri, p, q, od; pool_region(n, ri, p, q, od);
  int hs = p * HH / od, he = ((p + 1) * HH + od - 1) / od;
  int ws = q * WW / od, we = ((q + 1) * WW + od - 1) / od;
  const float* sp = src + bc * SS;
  float sum = 0.f;
  for (int hh = hs; hh < he; ++hh)
    for (int ww = ws; ww < we; ++ww) sum += sp[hh * WW + ww];
  dst[gid] = sum / (float)((he - hs) * (we - ws));
}

// ==================================================================
// 4) p + depthwise3x3(p) + bias (per-region, per-channel)
// ==================================================================
__global__ void dwadd_kernel(const float* __restrict__ raw,
                             const float* __restrict__ dcw,  // [4][C][1][3][3]
                             const float* __restrict__ dcb,  // [4][C]
                             float* __restrict__ fin) {
  size_t gid = (size_t)blockIdx.x * blockDim.x + threadIdx.x;
  if (gid >= (size_t)BB * CC * NPOOL) return;
  int n = (int)(gid % NPOOL);
  size_t bc = gid / NPOOL;
  int c = (int)(bc % CC);
  const int offs[4] = {0, 1024, 1280, 1401};
  int ri, p, q, od; pool_region(n, ri, p, q, od);
  const float* w9 = dcw + ((size_t)ri * CC + c) * 9;
  const float* base = raw + bc * NPOOL + offs[ri];
  float conv = dcb[ri * CC + c];
#pragma unroll
  for (int dy = -1; dy <= 1; ++dy)
#pragma unroll
    for (int dx = -1; dx <= 1; ++dx) {
      int pp = p + dy, qq = q + dx;
      if (pp >= 0 && pp < od && qq >= 0 && qq < od)
        conv += w9[(dy + 1) * 3 + (dx + 1)] * base[pp * od + qq];
    }
  fin[gid] = raw[gid] + conv;
}

// ==================================================================
// 5) LayerNorm over channels + repack to f16 [b][h][n(NPAD)][dh]
// ==================================================================
__global__ void ln_pack_kernel(const float* __restrict__ fin_k,
                               const float* __restrict__ fin_v,
                               const float* __restrict__ nw,
                               const float* __restrict__ nb,
                               _Float16* __restrict__ kf,
                               _Float16* __restrict__ vf) {
  __shared__ float red[CC];
  int n = blockIdx.x, b = blockIdx.y, which = blockIdx.z;
  const float* fin = which ? fin_v : fin_k;
  _Float16* out = which ? vf : kf;
  int c = threadIdx.x;
  float val = fin[((size_t)b * CC + c) * NPOOL + n];
  red[c] = val; __syncthreads();
  for (int st = CC / 2; st > 0; st >>= 1) { if (c < st) red[c] += red[c + st]; __syncthreads(); }
  float mu = red[0] / CC; __syncthreads();
  float d = val - mu;
  red[c] = d * d; __syncthreads();
  for (int st = CC / 2; st > 0; st >>= 1) { if (c < st) red[c] += red[c + st]; __syncthreads(); }
  float var = red[0] / CC;
  float y = d * rsqrtf(var + EPSF) * nw[c] + nb[c];
  int head = c >> 5, dh = c & 31;
  out[(((size_t)b * HEADS + head) * NPAD + n) * DH + dh] = (_Float16)y;
}

// ==================================================================
// 6) flash attention, one wave per (b,h,16-query tile).
//    K/V chunks (32 keys x 32 dh f16 = 2KB each) are staged to LDS with
//    double-buffered global_load_async_to_lds_b128 (ASYNCcnt), overlapping
//    the next chunk's DMA with the current chunk's 4 WMMAs.
//    Dynamic-LDS layout (halves): P[0,512) K0[512,1536) K1[1536,2560)
//                                 V0[2560,3584) V1[3584,4608)  => 9216 B
// ==================================================================
__global__ void attn_kernel(const _Float16* __restrict__ qf,
                            const _Float16* __restrict__ kf,
                            const _Float16* __restrict__ vf,
                            float* __restrict__ out_att) {   // [B][H][QTOT][DH]
  extern __shared__ _Float16 smem[];
  const int lane = threadIdx.x & 31;
  const int half = lane >> 4, lm = lane & 15;
  const int q0 = blockIdx.x * 16;
  const size_t bh = (size_t)blockIdx.z * HEADS + blockIdx.y;
  const _Float16* qbase = qf + (bh * QPAD + q0) * DH;
  const _Float16* kbase = kf + bh * (size_t)NPAD * DH;
  const _Float16* vbase = vf + bh * (size_t)NPAD * DH;

  // async-stage one 32-key chunk of K and V into LDS buffer `buf`
  auto issue_chunk = [&](int ck, int buf) {
    const char* kg = (const char*)(kbase + (size_t)ck * 32 * DH);
    const char* vg = (const char*)(vbase + (size_t)ck * 32 * DH);
    unsigned klo = (unsigned)((512 + buf * 1024) * 2);   // bytes
    unsigned vlo = (unsigned)((2560 + buf * 1024) * 2);  // bytes
#pragma unroll
    for (int i = 0; i < 4; ++i) {
      unsigned o = (unsigned)(i * 512 + lane * 16);
      asm volatile("global_load_async_to_lds_b128 %0, %1, off"
                   :: "v"(klo + o), "v"(kg + o) : "memory");
      asm volatile("global_load_async_to_lds_b128 %0, %1, off"
                   :: "v"(vlo + o), "v"(vg + o) : "memory");
    }
  };

  // Q fragment (contiguous by construction of the q layout)
  v16h qa;
  {
    const _Float16* qr = qbase + (size_t)lm * DH + half * 16;
#pragma unroll
    for (int j = 0; j < 8; ++j) { qa[j] = qr[j]; qa[8 + j] = qr[8 + j]; }
  }

  float rmax[8], rsum[8];
#pragma unroll
  for (int i = 0; i < 8; ++i) { rmax[i] = -3.0e38f; rsum[i] = 0.f; }
  v8f acc0 = {}, acc1 = {};

  issue_chunk(0, 0);
  for (int ck = 0; ck < NKCHUNK; ++ck) {
    if (ck + 1 < NKCHUNK) {
      issue_chunk(ck + 1, (ck + 1) & 1);
      asm volatile("s_wait_asynccnt 0x8" ::: "memory");  // chunk ck finished
    } else {
      asm volatile("s_wait_asynccnt 0x0" ::: "memory");
    }
    const _Float16* kl = smem + 512 + (ck & 1) * 1024;
    const _Float16* vl = smem + 2560 + (ck & 1) * 1024;
    const int k0 = ck * 32;

    // ---- scores for two 16-key tiles (pad keys hold zeros; masked below) ----
    v8f s01[2];
#pragma unroll
    for (int t = 0; t < 2; ++t) {
      v16h kb;
      const _Float16* kr = kl + (t * 16 + lm) * 32 + half * 16;
#pragma unroll
      for (int j = 0; j < 16; ++j) kb[j] = kr[j];
      v8f z = {};
      s01[t] = wmma_f16(qa, kb, z);
    }
    const bool v0 = (k0 + lm) < NPOOL, v1 = (k0 + 16 + lm) < NPOOL;

    // ---- online softmax (row stats live in frag index i) ----
    float p0v[8], p1v[8];
#pragma unroll
    for (int i = 0; i < 8; ++i) {
      float s0 = v0 ? s01[0][i] * SCALEF : -3.0e38f;
      float s1 = v1 ? s01[1][i] * SCALEF : -3.0e38f;
      float mx = fmaxf(s0, s1);
#pragma unroll
      for (int w = 1; w < 16; w <<= 1) mx = fmaxf(mx, __shfl_xor(mx, w, 32));
      float newmax = fmaxf(rmax[i], mx);
      float resc = __expf(rmax[i] - newmax);
      rmax[i] = newmax;
      float p0 = v0 ? __expf(s0 - newmax) : 0.f;
      float p1 = v1 ? __expf(s1 - newmax) : 0.f;
      float ps = p0 + p1;
#pragma unroll
      for (int w = 1; w < 16; w <<= 1) ps += __shfl_xor(ps, w, 32);
      rsum[i] = rsum[i] * resc + ps;
      acc0[i] *= resc; acc1[i] *= resc;
      p0v[i] = p0; p1v[i] = p1;
    }
    // ---- transpose P through LDS into an A-fragment ----
#pragma unroll
    for (int i = 0; i < 8; ++i) {
      int m = half * 8 + i;
      smem[m * 32 + lm]      = (_Float16)p0v[i];
      smem[m * 32 + 16 + lm] = (_Float16)p1v[i];
    }
    __syncthreads();
    v16h pa;
    {
      const _Float16* pr = smem + lm * 32 + half * 16;
#pragma unroll
      for (int j = 0; j < 8; ++j) { pa[j] = pr[j]; pa[8 + j] = pr[8 + j]; }
    }
    // ---- V fragments from LDS (two 16-wide dh halves) ----
    v16h vb0, vb1;
#pragma unroll
    for (int j = 0; j < 16; ++j) {
      const _Float16* vr = vl + (half * 16 + j) * 32;
      vb0[j] = vr[lm];
      vb1[j] = vr[16 + lm];
    }
    acc0 = wmma_f16(pa, vb0, acc0);
    acc1 = wmma_f16(pa, vb1, acc1);
    __syncthreads();
  }
#pragma unroll
  for (int i = 0; i < 8; ++i) {
    int qidx = q0 + half * 8 + i;
    if (qidx < QTOT) {
      float inv = 1.f / rsum[i];
      float* orow = out_att + (bh * QTOT + qidx) * DH;
      orow[lm]      = acc0[i] * inv;
      orow[16 + lm] = acc1[i] * inv;
    }
  }
}

// ==================================================================
// 7) w_attn: size-1 LN is exactly ln1_b -> gelu -> gqk rowsum -> softmax
// ==================================================================
__global__ void wattn_kernel(const float* __restrict__ ln1_b,
                             const float* __restrict__ gqk_w,
                             const float* __restrict__ gqk_b,
                             _Float16* __restrict__ wattn) { // [H][DH][DH]
  __shared__ float wvals[2 * CC];
  int o = threadIdx.x;                 // 512 threads
  float lb = ln1_b[0];
  float g = 0.5f * lb * (1.0f + erff(lb * 0.70710678118f));  // exact gelu
  float rs = 0.f;
  for (int d = 0; d < CC; ++d) rs += gqk_w[(size_t)o * CC + d];
  wvals[o] = g * rs + gqk_b[o];
  __syncthreads();
  for (int idx = threadIdx.x; idx < HEADS * DH * DH; idx += blockDim.x) {
    int h = idx >> 10, n = (idx >> 5) & 31, j = idx & 31;
    float wq = wvals[h * 64 + n] * SCALEF;
    float mx = -3.0e38f, sum = 0.f;
    for (int jj = 0; jj < DH; ++jj) mx = fmaxf(mx, wq * wvals[h * 64 + 32 + jj]);
    for (int jj = 0; jj < DH; ++jj) sum += __expf(wq * wvals[h * 64 + 32 + jj] - mx);
    wattn[idx] = (_Float16)(__expf(wq * wvals[h * 64 + 32 + j] - mx) / sum);
  }
}

// ==================================================================
// 8) agg = att_fea @ w_attn, scattered to channel layout c = w*HEADS+head
// ==================================================================
__global__ void agg_kernel(const float* __restrict__ out_att,
                           const _Float16* __restrict__ wattn,
                           _Float16* __restrict__ pre) {      // [B][C][S] f16
  const int lane = threadIdx.x & 31;
  const int half = lane >> 4, lm = lane & 15;
  const int q0 = blockIdx.x * 16;
  const int h = blockIdx.y, b = blockIdx.z;
  const size_t bh = (size_t)b * HEADS + h;
  v16h a;
  {
    const float* ar = out_att + (bh * QTOT + 1 + q0 + lm) * DH + half * 16;
#pragma unroll
    for (int j = 0; j < 8; ++j) { a[j] = (_Float16)ar[j]; a[8 + j] = (_Float16)ar[8 + j]; }
  }
  const _Float16* wb = wattn + (size_t)h * DH * DH;
  v16h b0, b1;
#pragma unroll
  for (int j = 0; j < 16; ++j) {
    int k = half * 16 + j;
    b0[j] = wb[k * DH + lm];
    b1[j] = wb[k * DH + 16 + lm];
  }
  v8f z0 = {}, z1 = {};
  v8f o0 = wmma_f16(a, b0, z0);
  v8f o1 = wmma_f16(a, b1, z1);
#pragma unroll
  for (int i = 0; i < 8; ++i) {
    int s = q0 + half * 8 + i;
    pre[((size_t)b * CC + (lm * HEADS + h)) * SS + s]        = (_Float16)o0[i];
    pre[((size_t)b * CC + ((16 + lm) * HEADS + h)) * SS + s] = (_Float16)o1[i];
  }
}

// ==================================================================
// 9) output projection: 64(M) x 16(N) strip per wave, 4x B reuse
// ==================================================================
__global__ void proj_gemm_kernel(const _Float16* __restrict__ pre,
                                 const float* __restrict__ proj_w,
                                 const float* __restrict__ proj_b,
                                 float* __restrict__ out) {
  const int lane = threadIdx.x & 31;
  const int half = lane >> 4, lm = lane & 15;
  const int n0 = blockIdx.x * 16;
  const int m0 = blockIdx.y * 64;
  const int b  = blockIdx.z;
  const _Float16* pb = pre + (size_t)b * CC * SS;
  const float* wrow[4];
#pragma unroll
  for (int t = 0; t < 4; ++t) wrow[t] = proj_w + (size_t)(m0 + t * 16 + lm) * CC;

  v8f acc[4] = {{}, {}, {}, {}};
  for (int kk = 0; kk < CC; kk += 32) {
    v16h bf;
#pragma unroll
    for (int j = 0; j < 16; ++j)
      bf[j] = pb[(size_t)(kk + half * 16 + j) * SS + (n0 + lm)];
#pragma unroll
    for (int t = 0; t < 4; ++t) {
      v16h a;
#pragma unroll
      for (int i = 0; i < 8; ++i) {
        int kb = half * 16 + i * 2;
        a[2 * i]     = (_Float16)wrow[t][kk + kb];
        a[2 * i + 1] = (_Float16)wrow[t][kk + kb + 1];
      }
      acc[t] = wmma_f16(a, bf, acc[t]);
    }
  }
#pragma unroll
  for (int t = 0; t < 4; ++t) {
#pragma unroll
    for (int i = 0; i < 8; ++i) {
      int o = m0 + t * 16 + half * 8 + i;
      out[((size_t)b * CC + o) * SS + (n0 + lm)] = acc[t][i] + proj_b[o];
    }
  }
}

// ==================================================================
extern "C" void kernel_launch(void* const* d_in, const int* in_sizes, int n_in,
                              void* d_out, int out_size, void* d_ws, size_t ws_size,
                              hipStream_t stream) {
  const float* x      = (const float*)d_in[0];
  const float* q_w    = (const float*)d_in[1];
  const float* kv_w   = (const float*)d_in[2];
  const float* proj_w = (const float*)d_in[3];
  const float* proj_b = (const float*)d_in[4];
  const float* norm_w = (const float*)d_in[5];
  const float* norm_b = (const float*)d_in[6];
  const float* gtok   = (const float*)d_in[7];
  const float* ln1_b  = (const float*)d_in[9];
  const float* gqk_w  = (const float*)d_in[10];
  const float* gqk_b  = (const float*)d_in[11];
  const float* dconvw = (const float*)d_in[12];
  const float* dconvb = (const float*)d_in[13];
  float* out = (float*)d_out;

  char* ws = (char*)d_ws;
  size_t off = 0;
  auto alloc = [&](size_t bytes) -> void* {
    void* p = ws + off;
    off += (bytes + 255) & ~(size_t)255;
    return p;
  };
  _Float16* qf     = (_Float16*)alloc((size_t)BB * HEADS * QPAD * DH * 2);
  float*    k_map  = (float*)   alloc((size_t)BB * CC * SS * 4);
  float*    v_map  = (float*)   alloc((size_t)BB * CC * SS * 4);
  float*    praw_k = (float*)   alloc((size_t)BB * CC * NPOOL * 4);
  float*    praw_v = (float*)   alloc((size_t)BB * CC * NPOOL * 4);
  float*    pfin_k = (float*)   alloc((size_t)BB * CC * NPOOL * 4);
  float*    pfin_v = (float*)   alloc((size_t)BB * CC * NPOOL * 4);
  _Float16* kf     = (_Float16*)alloc((size_t)BB * HEADS * NPAD * DH * 2);
  _Float16* vf     = (_Float16*)alloc((size_t)BB * HEADS * NPAD * DH * 2);
  float*    oatt   = (float*)   alloc((size_t)BB * HEADS * QTOT * DH * 4);
  _Float16* wattn  = (_Float16*)alloc((size_t)HEADS * DH * DH * 2);
  _Float16* pre    = (_Float16*)alloc((size_t)BB * CC * SS * 2);

  qinit_kernel<<<8, 256, 0, stream>>>(gtok, qf);
  kvpad_kernel<<<112, 256, 0, stream>>>(kf, vf);

  qkv_gemm_kernel<<<dim3(SS / 16, 768 / 64, BB), 32, 0, stream>>>(
      x, q_w, kv_w, qf, k_map, v_map);

  const size_t npc = (size_t)BB * CC * NPOOL;
  const int pgrid = (int)((npc + 255) / 256);
  pool_kernel<<<pgrid, 256, 0, stream>>>(k_map, praw_k);
  pool_kernel<<<pgrid, 256, 0, stream>>>(v_map, praw_v);
  dwadd_kernel<<<pgrid, 256, 0, stream>>>(praw_k, dconvw, dconvb, pfin_k);
  dwadd_kernel<<<pgrid, 256, 0, stream>>>(praw_v, dconvw, dconvb, pfin_v);

  ln_pack_kernel<<<dim3(NPOOL, BB, 2), CC, 0, stream>>>(
      pfin_k, pfin_v, norm_w, norm_b, kf, vf);

  wattn_kernel<<<1, 512, 0, stream>>>(ln1_b, gqk_w, gqk_b, wattn);

  attn_kernel<<<dim3(QPAD / 16, HEADS, BB), 32, 9216, stream>>>(qf, kf, vf, oatt);

  agg_kernel<<<dim3(SS / 16, HEADS, BB), 32, 0, stream>>>(oatt, wattn, pre);

  proj_gemm_kernel<<<dim3(SS / 16, CC / 64, BB), 32, 0, stream>>>(
      pre, proj_w, proj_b, out);
}
